// SelfSupervisedVAE_39109972198210
// MI455X (gfx1250) — matrix operand order repaired and checked by reading, not compiled, not run on hardware
//
#include <hip/hip_runtime.h>
#include <hip/hip_bf16.h>

// ---------------------------------------------------------------------------
// SelfSupervisedVAE (context-gated LSTM + VAE head) for MI455X / gfx1250.
// B=256, S=128, F=128, H=512, C=64, L=16
//
// Pipeline (all on `stream`):
//   0) init            : zero h-buffer[0] and the grid-barrier counter
//   1) ctx_hidden      : relu(context @ cg_w1 + cg_b1)            [256,512]
//   2) ctx_gate        : sigmoid(hid @ cg_w2 + cg_b2)             [256,1536]
//   3) pack_wih        : W_ih f32 -> bf16 in WMMA-B fragment order
//   4) xg_gemm (WMMA)  : xg = bf16(x) @ bf16(W_ih) + (b_ih+b_hh), stored in
//                        WMMA-D tile layout [S][16 mt][128 nt][32 lane][8 r]
//   5) lstm (WMMA, persistent, 64 WGs, grid barrier per step):
//        gates = h@W_hh (LDS-staged bf16, padded rows) + xg_tile; c in VGPRs;
//        h kept in WMMA A-fragment layout so next-step A loads are contiguous
//   6) head_muz        : mu / log_var / z (reparameterize)
//   7) dec1, dec2      : relu(z@dec_w1+b) @ dec_w2 + b -> recon_x
// ---------------------------------------------------------------------------

#define B_ 256
#define S_ 128
#define F_ 128
#define H_ 512
#define C_ 64
#define L_ 16

// LDS row padding: 16 data elems padded to 24 (48B stride -> 16B-aligned b128,
// 2-way LDS bank access instead of 4-way).
#define WROW 24

typedef __attribute__((ext_vector_type(16))) __bf16 v16bf;
typedef __attribute__((ext_vector_type(8)))  __bf16 v8bf;
typedef __attribute__((ext_vector_type(8)))  float  v8f;

static __device__ __forceinline__ float fsig(float x) {
    return __builtin_amdgcn_rcpf(1.0f + __expf(-x));     // v_exp + v_rcp
}
static __device__ __forceinline__ float ftanh(float x) {
    float e = __expf(-2.0f * x);
    return (1.0f - e) * __builtin_amdgcn_rcpf(1.0f + e);
}

// ---------------- 0) init: zero h-buffer 0 (h0 = 0) and barrier counter -----
__global__ __launch_bounds__(256)
void k_init(float4* __restrict__ hb0_f4, unsigned* __restrict__ cnt) {
    int i = blockIdx.x * 256 + threadIdx.x;
    if (i < (B_ * H_ * 2 / 16)) {                // 256KB bf16 = 16384 float4
        hb0_f4[i] = make_float4(0.f, 0.f, 0.f, 0.f);
    }
    if (i == 0) *cnt = 0u;
}

// ---------------- 1) context hidden: relu(context @ cg_w1 + cg_b1) ----------
__global__ __launch_bounds__(256)
void k_ctx_hidden(const float* __restrict__ context, const float* __restrict__ w1,
                  const float* __restrict__ b1, float* __restrict__ hid) {
    int t = blockIdx.x * 256 + threadIdx.x;      // 256*512
    int b = t >> 9, j = t & (H_ - 1);
    float acc = b1[j];
    #pragma unroll 8
    for (int c = 0; c < C_; ++c) acc = fmaf(context[b * C_ + c], w1[c * H_ + j], acc);
    hid[t] = fmaxf(acc, 0.0f);
}

// ---------------- 2) context gates: sigmoid(hid @ cg_w2 + cg_b2) ------------
__global__ __launch_bounds__(256)
void k_ctx_gate(const float* __restrict__ hid, const float* __restrict__ w2,
                const float* __restrict__ b2, float* __restrict__ ctx) {
    int t = blockIdx.x * 256 + threadIdx.x;      // 256*1536
    int b = t / (3 * H_), j = t % (3 * H_);
    float acc = b2[j];
    #pragma unroll 8
    for (int k = 0; k < H_; ++k) acc = fmaf(hid[b * H_ + k], w2[k * 3 * H_ + j], acc);
    ctx[t] = fsig(acc);
}

// ---------------- 3) W_ih f32 -> bf16, WMMA-B fragment order ----------------
// wihp[((nt*4 + kc)*32 + lane)*16 + e] = bf16(W_ih[(kc*32+kk)*2048 + nt*16+(lane&15)]), kk = ((e&8)<<1)|((lane>>4)<<3)|(e&7)
// -> each (nt,kc) fragment read by xg_gemm is one contiguous, coalesced 1KB.
__global__ __launch_bounds__(256)
void k_pack_wih(const float* __restrict__ w, __bf16* __restrict__ wp) {
    int t = blockIdx.x * 256 + threadIdx.x;      // 128*2048 = 262144
    int e    = t & 15;
    int lane = (t >> 4) & 31;
    int kc   = (t >> 9) & 3;
    int nt   = t >> 11;
    wp[t] = (__bf16)w[(size_t)(kc * 32 + (((e & 8) << 1) | ((lane >> 4) << 3) | (e & 7))) * (4 * H_) + nt * 16 + (lane & 15)];
}

// ---------------- 4) xg = bf16(x) @ bf16(W_ih) + (b_ih + b_hh) --------------
// Output in WMMA-D tile order: xg[((s*16+mt)*128+nt)*256 + lane*8 + r]
// nt loop unrolled x2 with ping-pong fragment buffers (no register rotation).
#define XG_LOAD(BUF, NT)                                                        \
    _Pragma("unroll")                                                           \
    for (int kc = 0; kc < 4; ++kc)                                              \
        BUF[kc] = *(const v16bf*)(wp + (size_t)(((NT) * 4 + kc) * 32 + lane) * 16);

#define XG_STEP(NT, BUF) do {                                                   \
    float bb_ = b_ih[(NT) * 16 + ln16] + b_hh[(NT) * 16 + ln16];                \
    v8f acc_;                                                                   \
    _Pragma("unroll")                                                           \
    for (int i = 0; i < 8; ++i) acc_[i] = bb_;                                  \
    _Pragma("unroll")                                                           \
    for (int kc = 0; kc < 4; ++kc)                                              \
        acc_ = __builtin_amdgcn_wmma_f32_16x16x32_bf16(false, a[kc], false,     \
                   BUF[kc], (short)0, acc_, false, false);                      \
    *(v8f*)(xg + obase + (size_t)(NT) * 256) = acc_;                            \
} while (0)

__global__ __launch_bounds__(256)
void k_xg_gemm(const float* __restrict__ x, const __bf16* __restrict__ wp,
               const float* __restrict__ b_ih, const float* __restrict__ b_hh,
               float* __restrict__ xg) {
    const int wv   = blockIdx.x * 8 + (threadIdx.x >> 5);  // 0..2047
    const int lane = threadIdx.x & 31;
    const int ln16 = lane & 15;
    const int half = lane >> 4;
    const int s    = wv >> 4;          // time step
    const int mt   = wv & 15;          // batch tile

    // A fragments: rows b = mt*16 + ln16 of x[:, s, :]  (F = 128 -> 4 k-chunks)
    // ISA 16-bit A layout: elems 0..7 = K kc*32+half*8.. ; 8..15 = +16.
    const float* xrow = x + ((size_t)(mt * 16 + ln16) * S_ + s) * F_;
    v16bf a[4];
    #pragma unroll
    for (int kc = 0; kc < 4; ++kc) {
        v8f lof = *(const v8f*)(xrow + kc * 32 + half * 8);
        v8f hif = *(const v8f*)(xrow + kc * 32 + 16 + half * 8);
        v8bf lo, hi;
        #pragma unroll
        for (int i = 0; i < 8; ++i) { lo[i] = (__bf16)lof[i]; hi[i] = (__bf16)hif[i]; }
        a[kc] = __builtin_shufflevector(lo, hi, 0,1,2,3,4,5,6,7,8,9,10,11,12,13,14,15);
    }

    const size_t obase = (size_t)(s * 16 + mt) * 128 * 256 + (size_t)lane * 8;

    v16bf ba[4], bb[4];
    XG_LOAD(ba, 0);
    for (int nt = 0; nt < 128; nt += 2) {
        XG_LOAD(bb, nt + 1);
        XG_STEP(nt, ba);
        XG_LOAD(ba, (nt + 2) & 127);
        XG_STEP(nt + 1, bb);
    }
}

// ---------------- 5) persistent LSTM ----------------------------------------
// 64 WGs x 256 threads (8 waves). WG (mg,jt): mg = rows 0-127 / 128-255,
// jt = 16-column group of H. Wave w -> h-tile (mt = mg*8+w, jt).
// Each wave does 4 gate tiles x 16 k-chunks = 64 wmma per step; c stays in VGPRs.
//
// h buffers are stored in WMMA A-FRAGMENT layout:
//   h[((mt*16 + kc)*32 + lane)*16 + e]   (bf16, 256 KB)
// so each step's A load is one contiguous 1KB burst per (wave, kc).
// Writer mapping (from ISA A-tile table), for hidden column j = kc*32 + jj:
//   jj<8: half=0,e=jj ; 8..15: half=1,e=jj-8 ; 16..23: half=0,e=jj-8 ; 24..31: half=1,e=jj-16
//   => hb = (jj>>3)&1 ; e = (jj&7) | ((jj&16)>>1) ; lane' = (m&15) + 16*hb
__global__ __launch_bounds__(256)
void k_lstm(const float* __restrict__ W_hh, const float* __restrict__ ctx,
            const float* __restrict__ xg, __bf16* __restrict__ hb0,
            __bf16* __restrict__ hb1, float* __restrict__ hfin,
            unsigned* __restrict__ cnt) {
    __shared__ __bf16 ldsW[4 * H_ * WROW];   // 96 KB: [gate][kc*32+lane][24-padded B-frag row]

    const int tid  = threadIdx.x;
    const int wg   = blockIdx.x;           // 0..63
    const int mg   = wg >> 5;              // 0..1
    const int jt   = wg & 31;              // 0..31
    const int wave = tid >> 5;             // 0..7
    const int lane = tid & 31;
    const int ln16 = lane & 15;
    const int half = lane >> 4;
    const int mt   = mg * 8 + wave;        // 0..15
    const int m0   = mt * 16;
    const int j0   = jt * 16;

    // Stage this WG's W_hh slice into LDS (f32 -> bf16), padded rows.
    for (int i = tid; i < 4 * H_ * 16; i += 256) {
        int g = i >> 13;                   // /(512*16)
        int r = i & 8191;
        int k = r >> 4;
        int c = r & 15;
        ldsW[(g * H_ + (k & ~31) + c + 16 * ((k >> 3) & 1)) * WROW + ((k & 7) | ((k & 16) >> 1))] =
            (__bf16)W_hh[(size_t)k * (4 * H_) + g * H_ + j0 + c];
    }
    __syncthreads();

    // Static per-element context gates + cell state (VGPR-resident all 128 steps).
    float ci[8], cf[8], co[8], cc[8];
    int rowm[8];
    #pragma unroll
    for (int r = 0; r < 8; ++r) {
        int m = m0 + r + half * 8;         // ISA C/D layout: VGPR r -> M = r / r+8
        rowm[r] = m;
        ci[r] = ctx[(size_t)m * (3 * H_) + 0 * H_ + j0 + ln16];
        cf[r] = ctx[(size_t)m * (3 * H_) + 1 * H_ + j0 + ln16];
        co[r] = ctx[(size_t)m * (3 * H_) + 2 * H_ + j0 + ln16];
        cc[r] = 0.0f;
    }

    // Writer-side A-fragment mapping for this wave's h-tile columns.
    const int kcw = jt >> 1;                       // K-chunk of column j0+ln16
    const int jj  = (jt & 1) * 16 + ln16;          // column within the chunk
    const int hbw = (jj >> 3) & 1;
    const int ew  = (jj & 7) | ((jj & 16) >> 1);
    const size_t wfrag = (size_t)(mt * 16 + kcw) * 32 * 16;   // frag base (elems)

    for (int s = 0; s < S_; ++s) {
        const __bf16* hprev = (s & 1) ? hb1 : hb0;
        __bf16*       hnext = (s & 1) ? hb0 : hb1;

        // Accumulators initialized from precomputed xg tiles (bias folded in).
        const size_t xb = (size_t)(s * 16 + mt) * 128 * 256 + (size_t)lane * 8;
        v8f acc0 = *(const v8f*)(xg + xb + (size_t)(jt +  0) * 256);
        v8f acc1 = *(const v8f*)(xg + xb + (size_t)(jt + 32) * 256);
        v8f acc2 = *(const v8f*)(xg + xb + (size_t)(jt + 64) * 256);
        v8f acc3 = *(const v8f*)(xg + xb + (size_t)(jt + 96) * 256);

        // Prefetch next step's xg tiles into cache while we compute this step.
        if (s + 1 < S_) {
            const size_t xb2 = (size_t)((s + 1) * 16 + mt) * 128 * 256 + (size_t)lane * 8;
            __builtin_prefetch(xg + xb2 + (size_t)(jt +  0) * 256, 0, 0);
            __builtin_prefetch(xg + xb2 + (size_t)(jt + 32) * 256, 0, 0);
            __builtin_prefetch(xg + xb2 + (size_t)(jt + 64) * 256, 0, 0);
            __builtin_prefetch(xg + xb2 + (size_t)(jt + 96) * 256, 0, 0);
        }

        #pragma unroll
        for (int kc = 0; kc < 16; ++kc) {
            // A fragment: one contiguous 32B/lane load (h kept in frag layout).
            v16bf a = *(const v16bf*)(hprev + ((size_t)(mt * 16 + kc) * 32 + lane) * 16);
            const __bf16* bp = &ldsW[(kc * 32 + lane) * WROW];
            v16bf b0 = *(const v16bf*)(bp);
            v16bf b1 = *(const v16bf*)(bp + 1 * H_ * WROW);
            v16bf b2 = *(const v16bf*)(bp + 2 * H_ * WROW);
            v16bf b3 = *(const v16bf*)(bp + 3 * H_ * WROW);
            acc0 = __builtin_amdgcn_wmma_f32_16x16x32_bf16(false, a, false, b0, (short)0, acc0, false, false);
            acc1 = __builtin_amdgcn_wmma_f32_16x16x32_bf16(false, a, false, b1, (short)0, acc1, false, false);
            acc2 = __builtin_amdgcn_wmma_f32_16x16x32_bf16(false, a, false, b2, (short)0, acc2, false, false);
            acc3 = __builtin_amdgcn_wmma_f32_16x16x32_bf16(false, a, false, b3, (short)0, acc3, false, false);
        }

        // LSTM cell update (all operands local to this wave's tile).
        #pragma unroll
        for (int r = 0; r < 8; ++r) {
            float iv = fsig(acc0[r]) * ci[r];
            float fv = fsig(acc1[r]) * cf[r];
            float gv = ftanh(acc2[r]);
            float ov = fsig(acc3[r]) * co[r];
            float cn = fmaf(fv, cc[r], iv * gv);
            cc[r] = cn;
            float hv = ov * ftanh(cn);
            // Store into next h buffer in A-fragment layout.
            int lanep = (rowm[r] & 15) + 16 * hbw;
            hnext[wfrag + (size_t)lanep * 16 + ew] = (__bf16)hv;
            if (s == S_ - 1) {
                hfin[(size_t)rowm[r] * H_ + j0 + ln16] = hv;  // f32 for the head
            }
        }

        // Device-wide barrier (64 co-resident WGs): release h, acquire next.
        __threadfence();
        __syncthreads();
        if (tid == 0) {
            __hip_atomic_fetch_add(cnt, 1u, __ATOMIC_RELEASE, __HIP_MEMORY_SCOPE_AGENT);
            const unsigned target = 64u * (unsigned)(s + 1);
            while (__hip_atomic_load(cnt, __ATOMIC_ACQUIRE, __HIP_MEMORY_SCOPE_AGENT) < target) {
                __builtin_amdgcn_s_sleep(2);
            }
        }
        __syncthreads();
        __threadfence();
    }
}

// ---------------- 6) mu / log_var / z ---------------------------------------
__global__ __launch_bounds__(256)
void k_head_muz(const float* __restrict__ h, const float* __restrict__ mu_w,
                const float* __restrict__ mu_b, const float* __restrict__ lv_w,
                const float* __restrict__ lv_b, const float* __restrict__ eps,
                float* __restrict__ out, float* __restrict__ z) {
    int t = blockIdx.x * 256 + threadIdx.x;      // 256*16
    int b = t >> 4, l = t & 15;
    float mu = mu_b[l], lv = lv_b[l];
    const float* hrow = h + (size_t)b * H_;
    #pragma unroll 8
    for (int k = 0; k < H_; ++k) {
        float hf = hrow[k];
        mu = fmaf(hf, mu_w[k * L_ + l], mu);
        lv = fmaf(hf, lv_w[k * L_ + l], lv);
    }
    out[B_ * F_ + t]           = mu;             // mu at offset 32768
    out[B_ * F_ + B_ * L_ + t] = lv;             // log_var at offset 36864
    z[t] = fmaf(eps[t], __expf(0.5f * lv), mu);
}

// ---------------- 7a) decoder hidden ----------------------------------------
__global__ __launch_bounds__(256)
void k_dec1(const float* __restrict__ z, const float* __restrict__ w1,
            const float* __restrict__ b1, float* __restrict__ a1) {
    int t = blockIdx.x * 256 + threadIdx.x;      // 256*512
    int b = t >> 9, j = t & (H_ - 1);
    float acc = b1[j];
    #pragma unroll
    for (int l = 0; l < L_; ++l) acc = fmaf(z[b * L_ + l], w1[l * H_ + j], acc);
    a1[t] = fmaxf(acc, 0.0f);
}

// ---------------- 7b) decoder output ----------------------------------------
__global__ __launch_bounds__(256)
void k_dec2(const float* __restrict__ a1, const float* __restrict__ w2,
            const float* __restrict__ b2, float* __restrict__ out) {
    int t = blockIdx.x * 256 + threadIdx.x;      // 256*128
    int b = t >> 7, f = t & (F_ - 1);
    float acc = b2[f];
    #pragma unroll 8
    for (int k = 0; k < H_; ++k) acc = fmaf(a1[b * H_ + k], w2[k * F_ + f], acc);
    out[t] = acc;
}

// ---------------------------------------------------------------------------
extern "C" void kernel_launch(void* const* d_in, const int* in_sizes, int n_in,
                              void* d_out, int out_size, void* d_ws, size_t ws_size,
                              hipStream_t stream) {
    (void)in_sizes; (void)n_in; (void)out_size; (void)ws_size;
    const float* x       = (const float*)d_in[0];
    const float* context = (const float*)d_in[1];
    const float* eps     = (const float*)d_in[2];
    const float* W_ih    = (const float*)d_in[3];
    const float* b_ih    = (const float*)d_in[4];
    const float* W_hh    = (const float*)d_in[5];
    const float* b_hh    = (const float*)d_in[6];
    const float* cg_w1   = (const float*)d_in[7];
    const float* cg_b1   = (const float*)d_in[8];
    const float* cg_w2   = (const float*)d_in[9];
    const float* cg_b2   = (const float*)d_in[10];
    const float* mu_w    = (const float*)d_in[11];
    const float* mu_b    = (const float*)d_in[12];
    const float* lv_w    = (const float*)d_in[13];
    const float* lv_b    = (const float*)d_in[14];
    const float* dec_w1  = (const float*)d_in[15];
    const float* dec_b1  = (const float*)d_in[16];
    const float* dec_w2  = (const float*)d_in[17];
    const float* dec_b2  = (const float*)d_in[18];
    float* out = (float*)d_out;

    // Workspace layout.
    char* ws = (char*)d_ws;
    const size_t XG_OFF   = 0;                                  // 256 MiB f32
    const size_t HB_OFF   = XG_OFF   + (size_t)S_ * B_ * 4 * H_ * 4;
    const size_t HFIN_OFF = HB_OFF   + (size_t)2 * B_ * H_ * 2; // 2 bf16 h bufs
    const size_t CTX_OFF  = HFIN_OFF + (size_t)B_ * H_ * 4;     // f32 final h
    const size_t HID_OFF  = CTX_OFF  + (size_t)B_ * 3 * H_ * 4;
    const size_t WIH_OFF  = HID_OFF  + (size_t)B_ * H_ * 4;
    const size_t Z_OFF    = WIH_OFF  + (size_t)F_ * 4 * H_ * 2;
    const size_t A1_OFF   = Z_OFF    + (size_t)B_ * L_ * 4;
    const size_t CNT_OFF  = A1_OFF   + (size_t)B_ * H_ * 4;

    float*    xg   = (float*)(ws + XG_OFF);
    __bf16*   hb0  = (__bf16*)(ws + HB_OFF);
    __bf16*   hb1  = hb0 + (size_t)B_ * H_;
    float*    hfin = (float*)(ws + HFIN_OFF);
    float*    ctx  = (float*)(ws + CTX_OFF);
    float*    hid  = (float*)(ws + HID_OFF);
    __bf16*   wihp = (__bf16*)(ws + WIH_OFF);
    float*    z    = (float*)(ws + Z_OFF);
    float*    a1   = (float*)(ws + A1_OFF);
    unsigned* cnt  = (unsigned*)(ws + CNT_OFF);

    k_init<<<64, 256, 0, stream>>>((float4*)hb0, cnt);
    k_ctx_hidden<<<(B_ * H_) / 256, 256, 0, stream>>>(context, cg_w1, cg_b1, hid);
    k_ctx_gate<<<(B_ * 3 * H_) / 256, 256, 0, stream>>>(hid, cg_w2, cg_b2, ctx);
    k_pack_wih<<<(F_ * 4 * H_) / 256, 256, 0, stream>>>(W_ih, wihp);
    k_xg_gemm<<<256, 256, 0, stream>>>(x, wihp, b_ih, b_hh, xg);
    k_lstm<<<64, 256, 0, stream>>>(W_hh, ctx, xg, hb0, hb1, hfin, cnt);
    k_head_muz<<<(B_ * L_) / 256, 256, 0, stream>>>(hfin, mu_w, mu_b, lv_w, lv_b,
                                                    eps, out, z);
    k_dec1<<<(B_ * H_) / 256, 256, 0, stream>>>(z, dec_w1, dec_b1, a1);
    k_dec2<<<(B_ * F_) / 256, 256, 0, stream>>>(a1, dec_w2, dec_b2, out);
}
